// GCN_mhealth_1898375545326
// MI455X (gfx1250) — matrix operand order repaired and checked by reading, not compiled, not run on hardware
//
#include <hip/hip_runtime.h>
#include <hip/hip_bf16.h>

typedef _Float16 h16;
typedef __attribute__((ext_vector_type(16))) _Float16 v16h;
typedef __attribute__((ext_vector_type(8)))  _Float16 v8h;
typedef __attribute__((ext_vector_type(4)))  _Float16 v4h;
typedef __attribute__((ext_vector_type(8)))  float    v8f;

#define HID     128
#define INDIM   23
#define INPAD   32
#define NCLS    12
#define ROWT    4   // row tiles per block in the layer GEMM

// ---------------- WMMA fragment loaders ----------------

// A fragment: 16x32 f16 tile, rows row0..row0+15 of A (row-major, stride lda halves).
// lane<16: elements 0..7 = K kbase+0..7, 8..15 = K kbase+16..23; lane>=16: K offset +8.
__device__ __forceinline__ v16h load_a_frag(const h16* __restrict__ A, int lda,
                                            int row0, int kbase, int lane, int mMax) {
    int m = row0 + (lane & 15);
    if (m >= mMax) m = mMax - 1;             // clamp; junk rows masked at store
    int koff = (lane < 16) ? 0 : 8;
    const h16* p = A + (size_t)m * lda + kbase + koff;
    v8h lo = *(const v8h*)p;                 // 16B aligned
    v8h hi = *(const v8h*)(p + 16);
    v16h r;
#pragma unroll
    for (int i = 0; i < 8; ++i) { r[i] = lo[i]; r[i + 8] = hi[i]; }
    return r;
}

// B fragment from pre-swizzled weights: lane's 16 halves are contiguous (32B).
__device__ __forceinline__ v16h load_b_frag_sw(const h16* __restrict__ Wsw,
                                               int fragIdx, int lane) {
    const h16* p = Wsw + ((size_t)fragIdx * 32 + lane) * 16;
    v8h lo = *(const v8h*)p;
    v8h hi = *(const v8h*)(p + 8);
    v16h r;
#pragma unroll
    for (int i = 0; i < 8; ++i) { r[i] = lo[i]; r[i + 8] = hi[i]; }
    return r;
}

// ---------------- prep kernels ----------------

__global__ void k_pad_x(const float* __restrict__ x, h16* __restrict__ xp, int n) {
    int idx = blockIdx.x * blockDim.x + threadIdx.x;
    if (idx >= n * INPAD) return;
    int i = idx / INPAD, c = idx - i * INPAD;
    xp[idx] = (c < INDIM) ? (h16)x[i * INDIM + c] : (h16)0.0f;
}

// Swizzle f32 weight W (krows x ncols, row-major stride ldw) into WMMA-B fragment
// layout: frag f = kc*ctiles+ct; lane l owns 16 contiguous halves:
//   n = ct*16 + (l&15),  k = kc*32 + (l<16?0:8) + (e<8?e:e+8),  zero-padded out of range.
__global__ void k_swizzle_w(const float* __restrict__ W, h16* __restrict__ Wsw,
                            int kchunks, int ctiles, int ldw, int krows, int ncols) {
    int idx = blockIdx.x * blockDim.x + threadIdx.x;
    int total = kchunks * ctiles * 32 * 16;
    if (idx >= total) return;
    int e    = idx & 15;
    int lane = (idx >> 4) & 31;
    int f    = idx >> 9;
    int ct   = f % ctiles;
    int kc   = f / ctiles;
    int n = ct * 16 + (lane & 15);
    int k = kc * 32 + ((lane < 16) ? 0 : 8) + ((e < 8) ? e : e + 8);
    float v = (k < krows && n < ncols) ? W[(size_t)k * ldw + n] : 0.0f;
    Wsw[idx] = (h16)v;
}

// Fuse classifier: Wf = Wc1(128x64) @ Wc2(64x12), bf = bc1 @ Wc2 + bc2
__global__ void k_fuse_cls(const float* __restrict__ Wc1, const float* __restrict__ bc1,
                           const float* __restrict__ Wc2, const float* __restrict__ bc2,
                           float* __restrict__ Wf, float* __restrict__ bf) {
    int idx = blockIdx.x * blockDim.x + threadIdx.x;
    if (idx >= HID * NCLS) return;
    int r = idx / NCLS, c = idx - r * NCLS;
    float acc = 0.0f;
    for (int j = 0; j < 64; ++j) acc += Wc1[r * 64 + j] * Wc2[j * NCLS + c];
    Wf[idx] = acc;
    if (r == 0) {
        float bb = bc2[c];
        for (int j = 0; j < 64; ++j) bb += bc1[j] * Wc2[j * NCLS + c];
        bf[c] = bb;
    }
}

// ---------------- degree / norm ----------------

__global__ void k_deg_init(unsigned* __restrict__ deg, int n) {
    int i = blockIdx.x * blockDim.x + threadIdx.x;
    if (i < n) deg[i] = 1u;              // self loop
}

__global__ void k_deg_count(const int* __restrict__ dst, unsigned* __restrict__ deg, int E) {
    int e = blockIdx.x * blockDim.x + threadIdx.x;
    if (e < E) atomicAdd(&deg[dst[e]], 1u);
}

__global__ void k_dinv(const unsigned* __restrict__ deg, float* __restrict__ dinv, int n) {
    int i = blockIdx.x * blockDim.x + threadIdx.x;
    if (i < n) dinv[i] = rsqrtf((float)deg[i]);
}

// ---------------- WMMA GEMM: G[n x 128] = A[n x K] @ W[K x 128] (f16 in/out, f32 acc) ----

template <int KC>
__global__ void k_gemm_h(const h16* __restrict__ A, int lda,
                         const h16* __restrict__ Wsw,
                         h16* __restrict__ G, int nrows, int rowTiles) {
    const int lane = threadIdx.x & 31;
    const int wv   = threadIdx.x >> 5;          // 8 waves -> 8 col tiles
    v16h bfrag[KC];
#pragma unroll
    for (int kc = 0; kc < KC; ++kc)
        bfrag[kc] = load_b_frag_sw(Wsw, kc * 8 + wv, lane);

    for (int t = 0; t < ROWT; ++t) {
        int tile = blockIdx.x * ROWT + t;       // uniform across block
        if (tile >= rowTiles) return;
        int row0 = tile * 16;

        // prefetch next row tile's A rows (global_prefetch_b8)
        int ptile = tile + 1;
        if (ptile < rowTiles)
            __builtin_prefetch(A + (size_t)(ptile * 16 + (lane & 15)) * lda, 0, 0);

        v8f c = {0.f, 0.f, 0.f, 0.f, 0.f, 0.f, 0.f, 0.f};
#pragma unroll
        for (int kc = 0; kc < KC; ++kc) {
            v16h a = load_a_frag(A, lda, row0, kc * 32, lane, nrows);
            c = __builtin_amdgcn_wmma_f32_16x16x32_f16(false, a, false, bfrag[kc],
                                                       (short)0, c, false, false);
        }
        int n  = wv * 16 + (lane & 15);
        int mb = row0 + ((lane < 16) ? 0 : 8);
        h16* gp = G + (size_t)mb * HID + n;
        if (row0 + 16 <= nrows) {               // block-uniform fast path (always for N%16==0)
#pragma unroll
            for (int r = 0; r < 8; ++r) gp[(size_t)r * HID] = (h16)c[r];
        } else {
#pragma unroll
            for (int r = 0; r < 8; ++r)
                if (mb + r < nrows) gp[(size_t)r * HID] = (h16)c[r];
        }
    }
}

// Final GEMM: out[n x 12] = H[n x 128] @ Wfused[128 x 12] + bf, f32 output
__global__ void k_gemm_out(const h16* __restrict__ A, const h16* __restrict__ Wfsw,
                           const float* __restrict__ bf, float* __restrict__ out,
                           int nrows, int rowTiles) {
    const int lane = threadIdx.x & 31;
    int gw = (int)((blockIdx.x * blockDim.x + threadIdx.x) >> 5);
    if (gw >= rowTiles) return;                  // wave-uniform
    int row0 = gw * 16;
    v8f c = {0.f, 0.f, 0.f, 0.f, 0.f, 0.f, 0.f, 0.f};
#pragma unroll
    for (int kc = 0; kc < 4; ++kc) {
        v16h b = load_b_frag_sw(Wfsw, kc, lane);
        v16h a = load_a_frag(A, HID, row0, kc * 32, lane, nrows);
        c = __builtin_amdgcn_wmma_f32_16x16x32_f16(false, a, false, b,
                                                   (short)0, c, false, false);
    }
    int n  = lane & 15;
    int mb = row0 + ((lane < 16) ? 0 : 8);
    if (row0 + 16 <= nrows) {                    // wave-uniform fast path
        if (n < NCLS) {                          // single divergence point
            float  bb = bf[n];
            float* op = out + (size_t)mb * NCLS + n;
#pragma unroll
            for (int r = 0; r < 8; ++r) op[(size_t)r * NCLS] = c[r] + bb;
        }
    } else {
        if (n < NCLS) {
            float bb = bf[n];
#pragma unroll
            for (int r = 0; r < 8; ++r) {
                int m = mb + r;
                if (m < nrows) out[(size_t)m * NCLS + n] = c[r] + bb;
            }
        }
    }
}

// ---------------- edge aggregation: AGG[dst] += G[src] * dinv[src]*dinv[dst] ----------

__global__ void k_aggregate(const h16* __restrict__ G, const int* __restrict__ src,
                            const int* __restrict__ dst, const float* __restrict__ dinv,
                            float* __restrict__ AGG, int E) {
    long long tid = (long long)blockIdx.x * blockDim.x + threadIdx.x;
    int e    = (int)(tid >> 5);
    int lane = (int)(tid & 31);
    if (e >= E) return;
    int s = src[e], d = dst[e];
    float sc = dinv[s] * dinv[d];
    v4h g = *(const v4h*)(G + (size_t)s * HID + lane * 4);
    float* o = AGG + (size_t)d * HID + lane * 4;
#pragma unroll
    for (int i = 0; i < 4; ++i)
        unsafeAtomicAdd(o + i, (float)g[i] * sc);
}

// bias + self-loop term + tanh, write next-layer f16 activations
__global__ void k_bias_tanh(const float* __restrict__ AGG, const h16* __restrict__ G,
                            const float* __restrict__ dinv, const float* __restrict__ b,
                            h16* __restrict__ H, int n) {
    int idx = blockIdx.x * blockDim.x + threadIdx.x;
    if (idx >= n * HID) return;
    int i = idx >> 7, f = idx & 127;
    float di = dinv[i];
    float v  = AGG[idx] + di * di * (float)G[idx] + b[f];
    H[idx] = (h16)tanhf(v);
}

// ---------------- launcher ----------------

extern "C" void kernel_launch(void* const* d_in, const int* in_sizes, int n_in,
                              void* d_out, int out_size, void* d_ws, size_t ws_size,
                              hipStream_t stream) {
    const float* x    = (const float*)d_in[0];
    const int*   ei   = (const int*)d_in[1];
    const float* W1   = (const float*)d_in[2];
    const float* b1   = (const float*)d_in[3];
    const float* W2   = (const float*)d_in[4];
    const float* b2   = (const float*)d_in[5];
    const float* W3   = (const float*)d_in[6];
    const float* b3   = (const float*)d_in[7];
    const float* Wc1  = (const float*)d_in[8];
    const float* bc1  = (const float*)d_in[9];
    const float* Wc2  = (const float*)d_in[10];
    const float* bc2  = (const float*)d_in[11];

    const int N = in_sizes[0] / INDIM;
    const int E = in_sizes[1] / 2;
    const int* srcA = ei;        // edge_index[0]
    const int* dstA = ei + E;    // edge_index[1]

    // workspace carve-up (256B aligned)
    char* ws = (char*)d_ws;
    size_t off = 0;
    auto take = [&](size_t bytes) { size_t o = off; off = (off + bytes + 255) & ~(size_t)255; return o; };
    unsigned* deg   = (unsigned*)(ws + take((size_t)N * 4));
    float*    dinv  = (float*)   (ws + take((size_t)N * 4));
    h16*      Xp    = (h16*)     (ws + take((size_t)N * INPAD * 2));
    h16*      H     = (h16*)     (ws + take((size_t)N * HID * 2));
    h16*      G     = (h16*)     (ws + take((size_t)N * HID * 2));
    float*    AGG   = (float*)   (ws + take((size_t)N * HID * 4));
    h16*      W1sw  = (h16*)     (ws + take((size_t)1 * 8 * 32 * 16 * 2));
    h16*      W2sw  = (h16*)     (ws + take((size_t)4 * 8 * 32 * 16 * 2));
    h16*      W3sw  = (h16*)     (ws + take((size_t)4 * 8 * 32 * 16 * 2));
    float*    Wf    = (float*)   (ws + take((size_t)HID * NCLS * 4));
    h16*      Wfsw  = (h16*)     (ws + take((size_t)4 * 1 * 32 * 16 * 2));
    float*    bf    = (float*)   (ws + take((size_t)NCLS * 4));
    (void)ws_size; (void)n_in; (void)out_size;

    const int B = 256;
    auto blocks = [](long long work, int b) { return (int)((work + b - 1) / b); };

    // weights / input prep
    k_pad_x    <<<blocks((long long)N * INPAD, B), B, 0, stream>>>(x, Xp, N);
    k_swizzle_w<<<blocks(1 * 8 * 512, B), B, 0, stream>>>(W1, W1sw, 1, 8, HID, INDIM, HID);
    k_swizzle_w<<<blocks(4 * 8 * 512, B), B, 0, stream>>>(W2, W2sw, 4, 8, HID, HID, HID);
    k_swizzle_w<<<blocks(4 * 8 * 512, B), B, 0, stream>>>(W3, W3sw, 4, 8, HID, HID, HID);
    k_fuse_cls <<<blocks(HID * NCLS, B), B, 0, stream>>>(Wc1, bc1, Wc2, bc2, Wf, bf);
    k_swizzle_w<<<blocks(4 * 1 * 512, B), B, 0, stream>>>(Wf, Wfsw, 4, 1, NCLS, HID, NCLS);

    // degree / normalization
    k_deg_init <<<blocks(N, B), B, 0, stream>>>(deg, N);
    k_deg_count<<<blocks(E, B), B, 0, stream>>>(dstA, deg, E);
    k_dinv     <<<blocks(N, B), B, 0, stream>>>(deg, dinv, N);

    const int rowTiles  = (N + 15) / 16;
    const int gemmGrid  = (rowTiles + ROWT - 1) / ROWT;
    const int aggGrid   = blocks((long long)E * 32, B);
    const int actGrid   = blocks((long long)N * HID, B);
    const int outGrid   = blocks((long long)rowTiles * 32, B);

    // layer 1
    k_gemm_h<1><<<gemmGrid, B, 0, stream>>>(Xp, INPAD, W1sw, G, N, rowTiles);
    hipMemsetAsync(AGG, 0, (size_t)N * HID * 4, stream);
    k_aggregate<<<aggGrid, B, 0, stream>>>(G, srcA, dstA, dinv, AGG, E);
    k_bias_tanh<<<actGrid, B, 0, stream>>>(AGG, G, dinv, b1, H, N);

    // layer 2
    k_gemm_h<4><<<gemmGrid, B, 0, stream>>>(H, HID, W2sw, G, N, rowTiles);
    hipMemsetAsync(AGG, 0, (size_t)N * HID * 4, stream);
    k_aggregate<<<aggGrid, B, 0, stream>>>(G, srcA, dstA, dinv, AGG, E);
    k_bias_tanh<<<actGrid, B, 0, stream>>>(AGG, G, dinv, b2, H, N);

    // layer 3
    k_gemm_h<4><<<gemmGrid, B, 0, stream>>>(H, HID, W3sw, G, N, rowTiles);
    hipMemsetAsync(AGG, 0, (size_t)N * HID * 4, stream);
    k_aggregate<<<aggGrid, B, 0, stream>>>(G, srcA, dstA, dinv, AGG, E);
    k_bias_tanh<<<actGrid, B, 0, stream>>>(AGG, G, dinv, b3, H, N);

    // fused classifier
    k_gemm_out<<<outGrid, B, 0, stream>>>(H, Wfsw, bf, (float*)d_out, N, rowTiles);
}
